// LSTM_14061722927609
// MI455X (gfx1250) — compile-verified, matrix operand order checked by reading, not run on hardware
//
#include <hip/hip_runtime.h>
#include <hip/hip_bf16.h>

// ---------------- problem constants ----------------
#define S_LEN 512
#define BATCH 64
#define ISZ   512
#define HSZ   1024
#define NG    4096            // 4*HSZ (gate-major concatenated columns: n = g*H + h)
#define MS    32768           // BATCH*S_LEN

// recurrent kernel shape
#define RB_BLOCKS  64         // one block per h-tile (1024/16)
#define RB_THREADS 512        // 16 waves: wave = gate(2b) x m-tile(2b)
#define BKS        1040       // LDS B K-stride (bf16 elems): 32B-aligned, 2-way-balanced banks
#define GSTR       68         // LDS gates stride (floats): rows m and m+8 land 32 banks apart

typedef __attribute__((ext_vector_type(16))) __bf16 bf16x16;
typedef __attribute__((ext_vector_type(8)))  float  f32x8;

// ---------------- helpers ----------------
__device__ __forceinline__ unsigned short f2bf(float f) {
  union { float f; unsigned u; } v; v.f = f;
  unsigned u = v.u;
  u += 0x7FFFu + ((u >> 16) & 1u);   // round-to-nearest-even
  return (unsigned short)(u >> 16);
}

__device__ __forceinline__ float fsig(float x) {
  return 1.0f / (1.0f + __expf(-x));
}

// device-scope grid barrier (monotonic counter, zeroed once per launch)
__device__ __forceinline__ void grid_barrier(unsigned* bar, unsigned target) {
  __threadfence();                 // make this block's stores device-visible
  __syncthreads();
  if (threadIdx.x == 0) {
    __hip_atomic_fetch_add(bar, 1u, __ATOMIC_RELEASE, __HIP_MEMORY_SCOPE_AGENT);
    while (__hip_atomic_load(bar, __ATOMIC_ACQUIRE, __HIP_MEMORY_SCOPE_AGENT) < target) {
      __builtin_amdgcn_s_sleep(2);
    }
  }
  __syncthreads();
  __threadfence();
}

// ---------------- conversion kernels (fp32 -> bf16, weights transposed to [N][K]) ----------------
__global__ void conv_x_kernel(const float* __restrict__ x, unsigned short* __restrict__ xb, int n) {
  int i = blockIdx.x * blockDim.x + threadIdx.x;
  if (i < n) xb[i] = f2bf(x[i]);
}

__global__ void conv_wx_kernel(const float* __restrict__ wx, unsigned short* __restrict__ wxt) {
  int i = blockIdx.x * blockDim.x + threadIdx.x;      // i = n*ISZ + k
  if (i >= NG * ISZ) return;
  int k = i & (ISZ - 1);
  int n = i >> 9;                                     // / ISZ
  int g = n >> 10, h = n & (HSZ - 1);
  wxt[i] = f2bf(wx[((size_t)g * ISZ + k) * HSZ + h]); // W_x[g][k][h]
}

__global__ void conv_wh_kernel(const float* __restrict__ wh, unsigned short* __restrict__ wht) {
  int i = blockIdx.x * blockDim.x + threadIdx.x;      // i = n*HSZ + k
  if (i >= NG * HSZ) return;
  int k = i & (HSZ - 1);
  int n = i >> 10;                                    // / HSZ
  int g = n >> 10, h = n & (HSZ - 1);
  wht[i] = f2bf(wh[((size_t)g * HSZ + k) * HSZ + h]); // W_h[g][k][h]
}

// ---------------- phase 1: xg[s][b][n] = x[b][s][:] . W_x[:, n] + bias[n] ----------------
// one 16x16 output tile per wave, bf16 WMMA, K = 512 (16 wmma ops / tile)
__global__ __launch_bounds__(256)
void gemm_x_kernel(const unsigned short* __restrict__ xb,   // [MS][ISZ]  (row m = b*S + s)
                   const unsigned short* __restrict__ wxt,  // [NG][ISZ]
                   const float* __restrict__ bias,          // [NG]
                   float* __restrict__ xg)                  // [S][B][NG]
{
  const int lane = threadIdx.x & 31;
  const int wave = threadIdx.x >> 5;
  const int tile = blockIdx.x * 8 + wave;     // 524288 tiles total
  const int nt = tile & 255;                  // 256 n-tiles
  const int mt = tile >> 8;                   // 2048 m-tiles
  const int m0 = mt * 16, n0 = nt * 16;

  const int lrow  = lane & 15;
  const int khalf = (lane >> 4) * 16;

  const unsigned short* aptr = xb  + (size_t)(m0 + lrow) * ISZ + khalf;
  const unsigned short* bptr = wxt + (size_t)(n0 + lrow) * ISZ + khalf;

  f32x8 acc = {};
  #pragma unroll 4
  for (int k = 0; k < ISZ; k += 32) {
    bf16x16 af = *(const bf16x16*)(aptr + k);
    bf16x16 bf = *(const bf16x16*)(bptr + k);
    __builtin_prefetch(bptr + k + 128, 0, 1);   // global_prefetch_b8
    acc = __builtin_amdgcn_wmma_f32_16x16x32_bf16(false, af, false, bf,
                                                  (short)0, acc, false, false);
  }

  const int   col  = n0 + (lane & 15);
  const float bv   = bias[col];
  const int   rofs = (lane >> 4) * 8;           // lanes 16-31 hold rows M=8..15
  #pragma unroll
  for (int r = 0; r < 8; ++r) {
    int mg = m0 + r + rofs;                     // global row m = b*S + s
    int b  = mg >> 9;                           // / S_LEN
    int s  = mg & (S_LEN - 1);
    xg[(size_t)(s * BATCH + b) * NG + col] = acc[r] + bv;
  }
}

// ---------------- phase 2: persistent recurrent kernel ----------------
// block j owns h-columns [j*16, j*16+16) for ALL four gates -> gates stay in LDS,
// c lives in LDS for the whole sequence, ONE grid barrier per timestep.
// wave w: gate g = w>>2, m-tile mt = w&3. W_h slice (64 cols x 1024 K bf16) staged in LDS once.
__global__ __launch_bounds__(RB_THREADS)
void lstm_recurrent_kernel(const float* __restrict__ xg,           // [S][B][NG]
                           const unsigned short* __restrict__ wht, // [NG][HSZ] bf16
                           unsigned short* __restrict__ hbf,       // [2][B][HSZ] bf16 double buffer
                           unsigned* __restrict__ bar,
                           float* __restrict__ out)                // h then c, [2][B][HSZ]
{
  extern __shared__ char smem[];
  unsigned short* Blds  = (unsigned short*)smem;                    // [64][BKS] bf16
  float* gates_lds      = (float*)(smem + (size_t)64 * BKS * 2);    // [64][GSTR]
  float* c_lds          = gates_lds + 64 * GSTR;                    // [64][16]

  const int tid  = threadIdx.x;
  const int lane = tid & 31;
  const int wave = tid >> 5;          // 0..15
  const int g    = wave >> 2;         // gate 0..3
  const int mt   = wave & 3;          // m-tile 0..3
  const int j    = blockIdx.x;        // h-tile 0..63

  // ---- stage this block's W_h slice into LDS (once, reused 512 steps) ----
  // local col c = gg*16 + ch  ->  global col n = gg*1024 + j*16 + ch
  for (int idx = tid; idx < 64 * (HSZ / 8); idx += RB_THREADS) {    // 16B chunks
    int c  = idx >> 7;                 // / (1024/8)
    int ck = (idx & 127) * 8;          // k offset (elems)
    int gg = c >> 4, ch = c & 15;
    const unsigned short* src = wht + (size_t)(gg * HSZ + j * 16 + ch) * HSZ + ck;
    *(uint4*)(Blds + (size_t)c * BKS + ck) = *(const uint4*)src;
  }
  for (int e = tid; e < 64 * 16; e += RB_THREADS) c_lds[e] = 0.0f;  // c0 = 0
  __syncthreads();

  const int lrow  = lane & 15;
  const int khalf = (lane >> 4) * 16;
  const unsigned short* bld = Blds + (size_t)(g * 16 + lrow) * BKS + khalf;

  const int m0   = mt * 16;
  const int rofs = (lane >> 4) * 8;
  const int cloc = g * 16 + (lane & 15);               // local gate column in LDS
  const int ncol = g * HSZ + j * 16 + (lane & 15);     // global column for xg

  const unsigned nb = gridDim.x;
  unsigned phase = 0;

  #pragma unroll 1
  for (int t = 0; t < S_LEN; ++t) {
    // ---- gates_tile = h_prev @ W_h_slice  (A from global/L2, B from LDS) ----
    const unsigned short* hin   = hbf + (size_t)(t & 1) * (BATCH * HSZ);
    const unsigned short* abase = hin + (size_t)(m0 + lrow) * HSZ + khalf;

    f32x8 acc = {};
    #pragma unroll 4
    for (int k = 0; k < HSZ; k += 32) {
      bf16x16 af = *(const bf16x16*)(abase + k);
      bf16x16 bf = *(const bf16x16*)(bld + k);
      acc = __builtin_amdgcn_wmma_f32_16x16x32_bf16(false, af, false, bf,
                                                    (short)0, acc, false, false);
    }

    // add precomputed input gates, park pre-activations in LDS
    const float* xgt = xg + (size_t)t * (BATCH * NG);
    #pragma unroll
    for (int r = 0; r < 8; ++r) {
      int m = m0 + r + rofs;
      gates_lds[m * GSTR + cloc] = acc[r] + xgt[(size_t)m * NG + ncol];
      __builtin_prefetch(xgt + (size_t)(BATCH * NG) + (size_t)m * NG + ncol, 0, 1); // next step's xg
    }
    __syncthreads();

    // ---- block-local LSTM cell update: 1024 elems (64 m x 16 h), 2 per thread ----
    unsigned short* hout = hbf + (size_t)((t + 1) & 1) * (BATCH * HSZ);
    #pragma unroll
    for (int e = tid; e < 64 * 16; e += RB_THREADS) {
      int m = e >> 4, ch = e & 15;
      const float* gm = gates_lds + m * GSTR + ch;
      float ig  = fsig(gm[0]);
      float fg  = fsig(gm[16]);
      float gg2 = tanhf(gm[32]);
      float og  = fsig(gm[48]);
      float cp  = c_lds[e];
      float cn  = fg * cp + ig * gg2;
      float hn  = og * tanhf(cn);
      c_lds[e] = cn;
      int hcol = j * 16 + ch;
      hout[(size_t)m * HSZ + hcol] = f2bf(hn);
      if (t == S_LEN - 1) {
        out[(size_t)m * HSZ + hcol]              = hn;  // h_T
        out[BATCH * HSZ + (size_t)m * HSZ + hcol] = cn; // c_T
      }
    }

    // single grid barrier: h[t+1] visible everywhere; double buffer prevents RAW on h[t]
    ++phase; grid_barrier(bar, phase * nb);
  }
}

// ---------------- host launch ----------------
extern "C" void kernel_launch(void* const* d_in, const int* in_sizes, int n_in,
                              void* d_out, int out_size, void* d_ws, size_t ws_size,
                              hipStream_t stream) {
  (void)in_sizes; (void)n_in; (void)out_size; (void)ws_size;
  const float* x    = (const float*)d_in[0];   // [B,S,I]
  const float* W_x  = (const float*)d_in[1];   // [4,I,H]
  const float* W_h  = (const float*)d_in[2];   // [4,H,H]
  const float* bias = (const float*)d_in[3];   // [4,H]
  float* out = (float*)d_out;

  char* ws = (char*)d_ws;
  const size_t off_xg  = 0;                                   // 512 MB fp32
  const size_t off_xbf = off_xg  + (size_t)S_LEN*BATCH*NG*4;  // 32 MB bf16
  const size_t off_wxt = off_xbf + (size_t)MS*ISZ*2;          // 4 MB bf16 [N][K]
  const size_t off_wht = off_wxt + (size_t)NG*ISZ*2;          // 8 MB bf16 [N][K]
  const size_t off_hbf = off_wht + (size_t)NG*HSZ*2;          // 256 KB (h double buffer)
  const size_t off_bar = off_hbf + (size_t)2*BATCH*HSZ*2;     // barrier counter

  float*          xg  = (float*)(ws + off_xg);
  unsigned short* xbf = (unsigned short*)(ws + off_xbf);
  unsigned short* wxt = (unsigned short*)(ws + off_wxt);
  unsigned short* wht = (unsigned short*)(ws + off_wht);
  unsigned short* hbf = (unsigned short*)(ws + off_hbf);
  unsigned*       bar = (unsigned*)(ws + off_bar);

  // zero h0 (bf16 0 == 0x0000) and the barrier counter (graph-capture safe)
  hipMemsetAsync(hbf, 0, (size_t)2*BATCH*HSZ*2, stream);
  hipMemsetAsync(bar, 0, 256, stream);

  // fp32 -> bf16 conversions (weights transposed to [N][K] for contiguous B fragments)
  conv_x_kernel<<<(MS*ISZ + 255)/256, 256, 0, stream>>>(x, xbf, MS*ISZ);
  conv_wx_kernel<<<(NG*ISZ + 255)/256, 256, 0, stream>>>(W_x, wxt);
  conv_wh_kernel<<<(NG*HSZ + 255)/256, 256, 0, stream>>>(W_h, wht);

  // phase 1: big batched input GEMM (524288 tiles / 8 waves per block)
  gemm_x_kernel<<<65536, 256, 0, stream>>>(xbf, wxt, bias, xg);

  // phase 2: persistent recurrence, W_h + gates + c resident in WGP LDS
  const size_t smem = (size_t)64 * BKS * 2      // W_h slice (bf16)
                    + (size_t)64 * GSTR * 4     // gate pre-activations
                    + (size_t)64 * 16 * 4;      // cell state c
  lstm_recurrent_kernel<<<RB_BLOCKS, RB_THREADS, smem, stream>>>(xg, wht, hbf, bar, out);
}
